// GRU_29300266893722
// MI455X (gfx1250) — matrix operand order repaired and checked by reading, not compiled, not run on hardware
//
#include <hip/hip_runtime.h>

// ---------------------------------------------------------------------------
// GRU recurrence on MI455X (gfx1250, wave32, WMMA bf16 -> f32 accumulate).
//
// Strategy: fuse input projections into the recurrent GEMMs (K = 1024 + 256),
// keep all weights as bf16 [N][K] transposes resident in L2 (~8 MB), run one
// persistent kernel (32 WGs x 8 waves = 256 waves) where each wave owns one
// 16x16 output tile of the [64 x 1024] state matrix, with global sense-
// reversing barriers between the two dependent phases of each timestep.
// Activations use v_exp_f32 + v_rcp_f32 (no IEEE divide chains).
// ---------------------------------------------------------------------------

typedef __bf16 v16bf __attribute__((ext_vector_type(16)));
typedef __bf16 v8bf  __attribute__((ext_vector_type(8)));
typedef float  v8f   __attribute__((ext_vector_type(8)));

#define GRU_B   64
#define GRU_T   512
#define GRU_IN  256
#define GRU_H   1024
#define GRU_OUT 256
#define GRU_K   (GRU_H + GRU_IN)   // 1280 fused reduction
#define NWG     32                 // persistent workgroups
#define WAVES_PER_WG 8             // 256 threads / wave32

// ---- fragment loaders (contiguous k = 16*(lane/16)+s, same order in A and B) --------------------------------
// A (16x32 bf16): lanes 0-15 -> M=lane, K 0..15;
//                 lanes 16-31 -> M=lane-16, K 16..31.
// Per lane: two contiguous 8-element (16B) chunks of row M.
__device__ inline v16bf ldA(const __bf16* tile, int stride, int lane) {
    const __bf16* p = tile + (size_t)(lane & 15) * stride + ((lane >> 4) << 4);
    union { v16bf v; v8bf h[2]; } u;
    u.h[0] = *(const v8bf*)p;          // K = kb .. kb+7
    u.h[1] = *(const v8bf*)(p + 8);    // K = kb+8 .. kb+15
    return u.v;
}
// B (32x16 bf16), weights pre-transposed to [N][K] row-major:
// lanes 0-15 -> N=lane, K=0..15; lanes 16-31 -> N=lane-16, K=16..31.
// Per lane: 16 contiguous bf16 of one N-row.
__device__ inline v16bf ldB(const __bf16* tile, int stride, int lane) {
    const __bf16* p = tile + (size_t)(lane & 15) * stride + ((lane >> 4) << 4);
    union { v16bf v; v8bf h[2]; } u;
    u.h[0] = *(const v8bf*)p;
    u.h[1] = *(const v8bf*)(p + 8);
    return u.v;
}

__device__ inline v8f wmma_bf16(v16bf a, v16bf b, v8f c) {
    return __builtin_amdgcn_wmma_f32_16x16x32_bf16(
        false, a, false, b, (short)0, c, false, false);
}

// v_rcp_f32-based activations: one TRANS op instead of div_scale/div_fmas
// chains. ~1 ulp rcp error is far below the bf16 operand noise floor.
__device__ inline float fast_sigmoid(float x) {
    float e = __expf(-x);
    return __builtin_amdgcn_rcpf(1.0f + e);
}
__device__ inline float fast_tanh(float x) {
    float e = __expf(-2.0f * fabsf(x));
    float t = (1.0f - e) * __builtin_amdgcn_rcpf(1.0f + e);
    return copysignf(t, x);
}

// ---- device-scope sense-reversing grid barrier ----------------------------
__device__ inline void grid_barrier(unsigned* bar) {
    __threadfence();          // make this block's global stores device-visible
    __syncthreads();
    if (threadIdx.x == 0) {
        unsigned gen  = __hip_atomic_load(&bar[1], __ATOMIC_RELAXED,
                                          __HIP_MEMORY_SCOPE_AGENT);
        unsigned prev = __hip_atomic_fetch_add(&bar[0], 1u, __ATOMIC_ACQ_REL,
                                               __HIP_MEMORY_SCOPE_AGENT);
        if (prev == NWG - 1u) {
            __hip_atomic_store(&bar[0], 0u, __ATOMIC_RELAXED,
                               __HIP_MEMORY_SCOPE_AGENT);
            __hip_atomic_fetch_add(&bar[1], 1u, __ATOMIC_RELEASE,
                                   __HIP_MEMORY_SCOPE_AGENT);
        } else {
            while (__hip_atomic_load(&bar[1], __ATOMIC_ACQUIRE,
                                     __HIP_MEMORY_SCOPE_AGENT) == gen)
                __builtin_amdgcn_s_sleep(1);
        }
    }
    __syncthreads();
}

// ---------------------------------------------------------------------------
// Prep: bf16 weight transposes [N][K] (recurrent||input fused), bf16 X in
// [T][B][IN], zero state + barrier. Re-run every launch -> graph-replay safe.
// ---------------------------------------------------------------------------
__global__ void gru_prep(const float* __restrict__ X,
                         const float* __restrict__ Wx, const float* __restrict__ Wh,
                         const float* __restrict__ Uz, const float* __restrict__ Vz,
                         const float* __restrict__ Ur, const float* __restrict__ Vr,
                         const float* __restrict__ Wo,
                         unsigned* bar, float* h, float* zbuf,
                         __bf16* A_h, __bf16* A_hr,
                         __bf16* BWh, __bf16* BWz, __bf16* BWr,
                         __bf16* BWo, __bf16* Xbf) {
    size_t tid = (size_t)blockIdx.x * blockDim.x + threadIdx.x;
    size_t np  = (size_t)gridDim.x * blockDim.x;
    if (tid < 64) bar[tid] = 0u;
    for (size_t i = tid; i < (size_t)GRU_B * GRU_H; i += np) {
        h[i] = 0.0f; zbuf[i] = 0.0f;
        A_h[i] = (__bf16)0.0f; A_hr[i] = (__bf16)0.0f;
    }
    for (size_t i = tid; i < (size_t)GRU_H * GRU_K; i += np) {
        int n = (int)(i / GRU_K), k = (int)(i % GRU_K);
        if (k < GRU_H) {
            BWh[i] = (__bf16)Wh[(size_t)n * GRU_H + k];
            BWz[i] = (__bf16)Vz[(size_t)n * GRU_H + k];
            BWr[i] = (__bf16)Vr[(size_t)n * GRU_H + k];
        } else {
            int ki = k - GRU_H;
            BWh[i] = (__bf16)Wx[(size_t)n * GRU_IN + ki];
            BWz[i] = (__bf16)Uz[(size_t)n * GRU_IN + ki];
            BWr[i] = (__bf16)Ur[(size_t)n * GRU_IN + ki];
        }
    }
    for (size_t i = tid; i < (size_t)GRU_OUT * GRU_H; i += np)
        BWo[i] = (__bf16)Wo[i];
    for (size_t i = tid; i < (size_t)GRU_T * GRU_B * GRU_IN; i += np) {
        int t = (int)(i >> 14);          // / (64*256)
        int b = (int)(i >> 8) & 63;
        int c = (int)i & 255;
        Xbf[i] = (__bf16)X[((size_t)b * GRU_T + t) * GRU_IN + c];
    }
}

// ---------------------------------------------------------------------------
// Persistent recurrence kernel.
// ---------------------------------------------------------------------------
__launch_bounds__(256)
__global__ void gru_scan(const float* __restrict__ bx, const float* __restrict__ bz,
                         const float* __restrict__ br, const float* __restrict__ bo,
                         unsigned* bar, float* __restrict__ h, float* __restrict__ zbuf,
                         __bf16* __restrict__ A_h, __bf16* __restrict__ A_hr,
                         const __bf16* __restrict__ BWh, const __bf16* __restrict__ BWz,
                         const __bf16* __restrict__ BWr, const __bf16* __restrict__ BWo,
                         const __bf16* __restrict__ Xbf, float* __restrict__ out) {
    const int lane  = threadIdx.x & 31;
    const int gwave = blockIdx.x * WAVES_PER_WG + (threadIdx.x >> 5); // 0..255
    const int mi = gwave & 3;         // 4 M-tiles (B=64)
    const int ni = gwave >> 2;        // 64 N-tiles (H=1024)
    const int m0 = mi << 4;
    const int n0 = ni << 4;
    const int nC   = n0 + (lane & 15);            // C/D: N = lane%16
    const int mCb  = m0 + ((lane >> 4) << 3);     // C/D: M = 8*(lane/16)+v

    for (int t = 0; t < GRU_T; ++t) {
        const __bf16* Xt = Xbf + (size_t)t * GRU_B * GRU_IN;

        // ---- phase A: h_tilde = tanh(bx + [h*r | X_t] @ [Wh;Wx]^T); gate h ----
        {
            v8f acc = {};
            #pragma unroll 4
            for (int k0 = 0; k0 < GRU_K; k0 += 32) {
                v16bf a;
                if (k0 < GRU_H)
                    a = ldA(A_hr + (size_t)m0 * GRU_H + k0, GRU_H, lane);
                else
                    a = ldA(Xt + (size_t)m0 * GRU_IN + (k0 - GRU_H), GRU_IN, lane);
                v16bf b = ldB(BWh + (size_t)n0 * GRU_K + k0, GRU_K, lane);
                acc = wmma_bf16(a, b, acc);
            }
            float bias = bx[nC];
            #pragma unroll
            for (int v = 0; v < 8; ++v) {
                size_t idx = (size_t)(mCb + v) * GRU_H + nC;
                float ht   = fast_tanh(acc[v] + bias);
                float zo   = zbuf[idx];
                float hn   = zo * h[idx] + (1.0f - zo) * ht;
                h[idx]   = hn;
                A_h[idx] = (__bf16)hn;
            }
        }
        grid_barrier(bar);

        // ---- phase B: z = sig(bz + [h|X_t]@[Vz;Uz]^T), r = sig(... Vr;Ur) ----
        {
            v8f az = {}, ar = {};
            #pragma unroll 4
            for (int k0 = 0; k0 < GRU_K; k0 += 32) {
                v16bf a;
                if (k0 < GRU_H)
                    a = ldA(A_h + (size_t)m0 * GRU_H + k0, GRU_H, lane);
                else
                    a = ldA(Xt + (size_t)m0 * GRU_IN + (k0 - GRU_H), GRU_IN, lane);
                v16bf bzf = ldB(BWz + (size_t)n0 * GRU_K + k0, GRU_K, lane);
                v16bf brf = ldB(BWr + (size_t)n0 * GRU_K + k0, GRU_K, lane);
                az = wmma_bf16(a, bzf, az);
                ar = wmma_bf16(a, brf, ar);
            }
            float biasz = bz[nC], biasr = br[nC];
            #pragma unroll
            for (int v = 0; v < 8; ++v) {
                size_t idx = (size_t)(mCb + v) * GRU_H + nC;
                float zv = fast_sigmoid(az[v] + biasz);
                float rv = fast_sigmoid(ar[v] + biasr);
                zbuf[idx] = zv;
                A_hr[idx] = (__bf16)(rv * h[idx]);   // stage r⊙h for next step
            }
        }
        grid_barrier(bar);
    }

    // ---- output GEMM: out = h @ Wo^T + bo   (64 tiles -> waves 0..63) ----
    if (gwave < (GRU_B / 16) * (GRU_OUT / 16)) {
        const int omi = gwave & 3, oni = gwave >> 2;
        const int om0 = omi << 4, on0 = oni << 4;
        v8f acc = {};
        #pragma unroll 4
        for (int k0 = 0; k0 < GRU_H; k0 += 32) {
            v16bf a = ldA(A_h + (size_t)om0 * GRU_H + k0, GRU_H, lane);
            v16bf b = ldB(BWo + (size_t)on0 * GRU_H + k0, GRU_H, lane);
            acc = wmma_bf16(a, b, acc);
        }
        int on  = on0 + (lane & 15);
        int omb = om0 + ((lane >> 4) << 3);
        float bias = bo[on];
        #pragma unroll
        for (int v = 0; v < 8; ++v)
            out[(size_t)(omb + v) * GRU_OUT + on] = acc[v] + bias;
    }
}

// ---------------------------------------------------------------------------
extern "C" void kernel_launch(void* const* d_in, const int* in_sizes, int n_in,
                              void* d_out, int out_size, void* d_ws, size_t ws_size,
                              hipStream_t stream) {
    const float* X  = (const float*)d_in[0];
    const float* Wx = (const float*)d_in[1];
    const float* bx = (const float*)d_in[2];
    const float* Wh = (const float*)d_in[3];
    const float* Uz = (const float*)d_in[4];
    const float* bz = (const float*)d_in[5];
    const float* Vz = (const float*)d_in[6];
    const float* Ur = (const float*)d_in[7];
    const float* br = (const float*)d_in[8];
    const float* Vr = (const float*)d_in[9];
    const float* Wo = (const float*)d_in[10];
    const float* bo = (const float*)d_in[11];
    float* out = (float*)d_out;

    char* ws = (char*)d_ws;
    size_t off = 0;
    auto take = [&](size_t bytes) { char* p = ws + off; off += (bytes + 255) & ~(size_t)255; return p; };

    unsigned* bar  = (unsigned*)take(64 * sizeof(unsigned));
    float*    h    = (float*)   take((size_t)GRU_B * GRU_H * sizeof(float));
    float*    zbuf = (float*)   take((size_t)GRU_B * GRU_H * sizeof(float));
    __bf16*   A_h  = (__bf16*)  take((size_t)GRU_B * GRU_H * sizeof(__bf16));
    __bf16*   A_hr = (__bf16*)  take((size_t)GRU_B * GRU_H * sizeof(__bf16));
    __bf16*   BWh  = (__bf16*)  take((size_t)GRU_H * GRU_K * sizeof(__bf16));
    __bf16*   BWz  = (__bf16*)  take((size_t)GRU_H * GRU_K * sizeof(__bf16));
    __bf16*   BWr  = (__bf16*)  take((size_t)GRU_H * GRU_K * sizeof(__bf16));
    __bf16*   BWo  = (__bf16*)  take((size_t)GRU_OUT * GRU_H * sizeof(__bf16));
    __bf16*   Xbf  = (__bf16*)  take((size_t)GRU_T * GRU_B * GRU_IN * sizeof(__bf16));
    (void)ws_size; (void)in_sizes; (void)n_in; (void)out_size;

    gru_prep<<<512, 256, 0, stream>>>(X, Wx, Wh, Uz, Vz, Ur, Vr, Wo,
                                      bar, h, zbuf, A_h, A_hr,
                                      BWh, BWz, BWr, BWo, Xbf);
    gru_scan<<<NWG, 256, 0, stream>>>(bx, bz, br, bo,
                                      bar, h, zbuf, A_h, A_hr,
                                      BWh, BWz, BWr, BWo, Xbf, out);
}
